// MixtureOfExperts_28209345200699
// MI455X (gfx1250) — compile-verified, matrix operand order checked by reading, not run on hardware
//
#include <hip/hip_runtime.h>
#include <hip/hip_bf16.h>

#define DEV __device__ __forceinline__

typedef __bf16 bf16_t;
typedef __bf16 v16bf __attribute__((ext_vector_type(16)));
typedef float  v8f   __attribute__((ext_vector_type(8)));

namespace {
constexpr int kD   = 768;      // d_model
constexpr int kE   = 8;        // experts
constexpr int kH   = 768;      // per-expert hidden
constexpr int kHS  = 3072;     // shared hidden
constexpr int kN   = 4 * 1024; // tokens (B*S)
constexpr int kCh  = 128;      // hidden-column chunk (8 n-tiles)
}

// ---------------- helpers ----------------

DEV bf16_t f2bf(float f) {
  unsigned u = __builtin_bit_cast(unsigned, f);
  u += 0x7FFFu + ((u >> 16) & 1u);             // round-to-nearest-even
  unsigned short h = (unsigned short)(u >> 16);
  return __builtin_bit_cast(bf16_t, h);
}

// Load one lane's 16-bf16 WMMA fragment from a K-contiguous row.
// CDNA5 16-bit A/B layout: elements 0..7 = K..K+7, 8..15 = K+16..K+23.
DEV v16bf load_frag(const bf16_t* p) {
  union { uint4 u[2]; v16bf v; } t;
  t.u[0] = *reinterpret_cast<const uint4*>(p);
  t.u[1] = *reinterpret_cast<const uint4*>(p + 16);
  return t.v;
}

DEV v8f wmma_bf16(v16bf a, v16bf b, v8f c) {
  return __builtin_amdgcn_wmma_f32_16x16x32_bf16(false, a, false, b, (short)0, c,
                                                 false, false);
}

DEV float silu(float x) { return x / (1.f + __expf(-x)); }

// ---------------- small kernels ----------------

__global__ __launch_bounds__(256) void moe_zero_kernel(float* __restrict__ routed,
                                                       int n, int* __restrict__ counts) {
  int i = blockIdx.x * 256 + threadIdx.x;
  if (i < n) routed[i] = 0.f;
  if (i < kE) counts[i] = 0;
}

__global__ __launch_bounds__(256) void moe_convf2bf_kernel(const float* __restrict__ src,
                                                           bf16_t* __restrict__ dst,
                                                           long n) {
  for (long i = blockIdx.x * 256L + threadIdx.x; i < n; i += (long)gridDim.x * 256L)
    dst[i] = f2bf(src[i]);
}

// src: [slices][R][C] fp32; dst: [slices][C][R] bf16 (N-major rows, K contiguous)
__global__ __launch_bounds__(256) void moe_tconv_kernel(const float* __restrict__ src,
                                                        bf16_t* __restrict__ dst,
                                                        int R, int C, int slices) {
  long total = (long)R * C * slices;
  long rc = (long)R * C;
  for (long i = blockIdx.x * 256L + threadIdx.x; i < total; i += (long)gridDim.x * 256L) {
    long s = i / rc;
    long rem = i - s * rc;
    int r = (int)(rem / C), c = (int)(rem % C);
    dst[s * rc + (long)c * R + r] = f2bf(src[i]);
  }
}

// One wave32 per token: router softmax/top-2 + sigmoid gate.
__global__ __launch_bounds__(256) void moe_router_kernel(
    const float* __restrict__ x, const float* __restrict__ rw,
    const float* __restrict__ sgw, const float* __restrict__ sgb,
    int* __restrict__ idx, float* __restrict__ gates, int* __restrict__ rank,
    float* __restrict__ alpha, int* __restrict__ counts) {
  int tok = (blockIdx.x * 256 + threadIdx.x) >> 5;
  int lane = threadIdx.x & 31;
  if (tok >= kN) return;
  const float* xr = x + (long)tok * kD;
  float acc[kE] = {};
  float sg = 0.f;
  for (int d = lane; d < kD; d += 32) {
    float xv = xr[d];
    const float* r = rw + d * kE;
#pragma unroll
    for (int e = 0; e < kE; ++e) acc[e] += xv * r[e];
    sg += xv * sgw[d];
  }
#pragma unroll
  for (int off = 16; off > 0; off >>= 1) {
#pragma unroll
    for (int e = 0; e < kE; ++e) acc[e] += __shfl_xor(acc[e], off, 32);
    sg += __shfl_xor(sg, off, 32);
  }
  if (lane == 0) {
    float m = acc[0];
#pragma unroll
    for (int e = 1; e < kE; ++e) m = fmaxf(m, acc[e]);
    float p[kE];
#pragma unroll
    for (int e = 0; e < kE; ++e) p[e] = __expf(acc[e] - m);
    int i0 = 0;
#pragma unroll
    for (int e = 1; e < kE; ++e) if (p[e] > p[i0]) i0 = e;
    int i1 = (i0 == 0) ? 1 : 0;
#pragma unroll
    for (int e = 0; e < kE; ++e) if (e != i0 && p[e] > p[i1]) i1 = e;
    float g0 = p[i0], g1 = p[i1];
    float gs = g0 + g1;
    g0 /= gs; g1 /= gs;
    idx[tok * 2 + 0] = i0; idx[tok * 2 + 1] = i1;
    gates[tok * 2 + 0] = g0; gates[tok * 2 + 1] = g1;
    rank[tok * 2 + 0] = atomicAdd(&counts[i0], 1);
    rank[tok * 2 + 1] = atomicAdd(&counts[i1], 1);
    alpha[tok] = 1.f / (1.f + __expf(-(sg + sgb[0])));
  }
}

__global__ void moe_scan_kernel(const int* __restrict__ counts, int* __restrict__ bases) {
  if (threadIdx.x == 0 && blockIdx.x == 0) {
    int s = 0;
    for (int e = 0; e < kE; ++e) { bases[e] = s; s += counts[e]; }
  }
}

__global__ __launch_bounds__(256) void moe_scatter_kernel(
    const int* __restrict__ idx, const float* __restrict__ gates,
    const int* __restrict__ rank, const int* __restrict__ bases,
    int* __restrict__ perm, float* __restrict__ pgate) {
  int t = blockIdx.x * 256 + threadIdx.x;
  if (t >= kN) return;
#pragma unroll
  for (int j = 0; j < 2; ++j) {
    int e = idx[t * 2 + j];
    int slot = bases[e] + rank[t * 2 + j];
    perm[slot] = t;
    pgate[slot] = gates[t * 2 + j];
  }
}

// ---------------- routed experts ----------------
// 32-token M-tile (two 16-row halves) so every B fragment feeds 2 WMMAs.
// Hidden dim processed in 128-col chunks: act chunk in 8KB LDS, down-proj
// partials accumulate in persistent per-wave registers.
__global__ __launch_bounds__(256) void moe_expert_kernel(
    const bf16_t* __restrict__ xbf, const bf16_t* __restrict__ wgT,
    const bf16_t* __restrict__ wuT, const bf16_t* __restrict__ wdT,
    const int* __restrict__ counts, const int* __restrict__ bases,
    const int* __restrict__ perm, const float* __restrict__ pgate,
    float* __restrict__ routed) {
  __shared__ alignas(16) bf16_t xA[32 * kD];    // 48 KB
  __shared__ alignas(16) bf16_t actA[32 * kCh]; // 8 KB
  __shared__ int tokS[32];
  __shared__ float gS[32];

  const int e = blockIdx.x;
  const int Me = counts[e];
  const int base = bases[e];
  const int ntiles = (Me + 31) >> 5;
  const int wave = threadIdx.x >> 5;
  const int lane = threadIdx.x & 31;
  const int nlane = lane & 15;
  const int kbase = (lane < 16) ? 0 : 8;
  const int mbase = (lane < 16) ? 0 : 8;

  const bf16_t* wgE = wgT + (long)e * kH * kD;  // [H][D] K-contiguous
  const bf16_t* wuE = wuT + (long)e * kH * kD;
  const bf16_t* wdE = wdT + (long)e * kD * kH;  // [D][H] K-contiguous

  for (int tile = blockIdx.y; tile < ntiles; tile += gridDim.y) {
    if (threadIdx.x < 32) {
      int row = tile * 32 + threadIdx.x;
      int t = 0; float g = 0.f;
      if (row < Me) { t = perm[base + row]; g = pgate[base + row]; }
      tokS[threadIdx.x] = t; gS[threadIdx.x] = g;
    }
    __syncthreads();
    for (int i = threadIdx.x; i < 32 * kD; i += 256) {
      int r = i / kD, c = i - r * kD;
      xA[i] = xbf[(long)tokS[r] * kD + c];
    }
    __syncthreads();

    v8f acc0[6] = {}, acc1[6] = {};            // down-proj partials, 2 M-halves

    for (int ch = 0; ch < kH / kCh; ++ch) {
      // gate + up for this 128-col chunk; each wave owns one 16-col n-tile
      {
        int ncol = ch * kCh + wave * 16 + nlane;
        const bf16_t* bgp = wgE + (long)ncol * kD + kbase;
        const bf16_t* bup = wuE + (long)ncol * kD + kbase;
        const bf16_t* a0p = xA + (long)nlane * kD + kbase;
        const bf16_t* a1p = xA + (long)(16 + nlane) * kD + kbase;
        v8f cg0 = {}, cg1 = {}, cu0 = {}, cu1 = {};
        for (int kk = 0; kk < kD; kk += 32) {
          __builtin_prefetch(bgp + kk + 256, 0, 3);
          __builtin_prefetch(bup + kk + 256, 0, 3);
          v16bf bg = load_frag(bgp + kk);
          v16bf bu = load_frag(bup + kk);
          v16bf a0 = load_frag(a0p + kk);
          v16bf a1 = load_frag(a1p + kk);
          cg0 = wmma_bf16(a0, bg, cg0);
          cg1 = wmma_bf16(a1, bg, cg1);
          cu0 = wmma_bf16(a0, bu, cu0);
          cu1 = wmma_bf16(a1, bu, cu1);
        }
#pragma unroll
        for (int r = 0; r < 8; ++r) {
          actA[(mbase + r) * kCh + wave * 16 + nlane]      = f2bf(silu(cg0[r]) * cu0[r]);
          actA[(16 + mbase + r) * kCh + wave * 16 + nlane] = f2bf(silu(cg1[r]) * cu1[r]);
        }
      }
      __syncthreads();
      // down-proj partial over this chunk's 128 K values
      for (int j = 0; j < 6; ++j) {
        int nt = wave + j * 8;
        const bf16_t* bdp = wdE + (long)(nt * 16 + nlane) * kH + ch * kCh + kbase;
        const bf16_t* a0p = actA + (long)nlane * kCh + kbase;
        const bf16_t* a1p = actA + (long)(16 + nlane) * kCh + kbase;
        v8f c0 = acc0[j], c1 = acc1[j];
        for (int kk = 0; kk < kCh; kk += 32) {
          __builtin_prefetch(bdp + kk + 256, 0, 3);
          v16bf b = load_frag(bdp + kk);
          c0 = wmma_bf16(load_frag(a0p + kk), b, c0);
          c1 = wmma_bf16(load_frag(a1p + kk), b, c1);
        }
        acc0[j] = c0; acc1[j] = c1;
      }
      __syncthreads();
    }

    // gate-weighted scatter-add into routed accumulator
    for (int j = 0; j < 6; ++j) {
      int nt = wave + j * 8;
#pragma unroll
      for (int r = 0; r < 8; ++r) {
        int m0 = mbase + r;
        int m1 = 16 + mbase + r;
        atomicAdd(&routed[(long)tokS[m0] * kD + nt * 16 + nlane], acc0[j][r] * gS[m0]);
        atomicAdd(&routed[(long)tokS[m1] * kD + nt * 16 + nlane], acc1[j][r] * gS[m1]);
      }
    }
    __syncthreads();
  }
}

// ---------------- shared expert + blend ----------------
__global__ __launch_bounds__(256) void moe_shared_kernel(
    const bf16_t* __restrict__ xbf, const bf16_t* __restrict__ wsgT,
    const bf16_t* __restrict__ wsuT, const bf16_t* __restrict__ wsdT,
    const float* __restrict__ alpha, const float* __restrict__ routed,
    float* __restrict__ out) {
  __shared__ alignas(16) bf16_t xA[32 * kD];    // 48 KB
  __shared__ alignas(16) bf16_t actA[32 * kCh]; // 8 KB
  __shared__ float aS[32];

  const int tile = blockIdx.x;                  // kN/32 = 128 tiles
  const int wave = threadIdx.x >> 5;
  const int lane = threadIdx.x & 31;
  const int nlane = lane & 15;
  const int kbase = (lane < 16) ? 0 : 8;
  const int mbase = (lane < 16) ? 0 : 8;

  if (threadIdx.x < 32) aS[threadIdx.x] = alpha[tile * 32 + threadIdx.x];
  for (int i = threadIdx.x; i < 32 * kD; i += 256) {
    int r = i / kD, c = i - r * kD;
    xA[i] = xbf[(long)(tile * 32 + r) * kD + c];
  }
  __syncthreads();

  v8f acc0[6] = {}, acc1[6] = {};

  for (int ch = 0; ch < kHS / kCh; ++ch) {      // 24 chunks of 128 hidden cols
    {
      int ncol = ch * kCh + wave * 16 + nlane;  // row of transposed weight
      const bf16_t* bgp = wsgT + (long)ncol * kD + kbase;
      const bf16_t* bup = wsuT + (long)ncol * kD + kbase;
      const bf16_t* a0p = xA + (long)nlane * kD + kbase;
      const bf16_t* a1p = xA + (long)(16 + nlane) * kD + kbase;
      v8f cg0 = {}, cg1 = {}, cu0 = {}, cu1 = {};
      for (int kk = 0; kk < kD; kk += 32) {
        __builtin_prefetch(bgp + kk + 256, 0, 3);
        __builtin_prefetch(bup + kk + 256, 0, 3);
        v16bf bg = load_frag(bgp + kk);
        v16bf bu = load_frag(bup + kk);
        v16bf a0 = load_frag(a0p + kk);
        v16bf a1 = load_frag(a1p + kk);
        cg0 = wmma_bf16(a0, bg, cg0);
        cg1 = wmma_bf16(a1, bg, cg1);
        cu0 = wmma_bf16(a0, bu, cu0);
        cu1 = wmma_bf16(a1, bu, cu1);
      }
#pragma unroll
      for (int r = 0; r < 8; ++r) {
        actA[(mbase + r) * kCh + wave * 16 + nlane]      = f2bf(silu(cg0[r]) * cu0[r]);
        actA[(16 + mbase + r) * kCh + wave * 16 + nlane] = f2bf(silu(cg1[r]) * cu1[r]);
      }
    }
    __syncthreads();
    for (int j = 0; j < 6; ++j) {
      int nt = wave + j * 8;
      const bf16_t* bdp = wsdT + (long)(nt * 16 + nlane) * kHS + ch * kCh + kbase;
      const bf16_t* a0p = actA + (long)nlane * kCh + kbase;
      const bf16_t* a1p = actA + (long)(16 + nlane) * kCh + kbase;
      v8f c0 = acc0[j], c1 = acc1[j];
      for (int kk = 0; kk < kCh; kk += 32) {
        __builtin_prefetch(bdp + kk + 256, 0, 3);
        v16bf b = load_frag(bdp + kk);
        c0 = wmma_bf16(load_frag(a0p + kk), b, c0);
        c1 = wmma_bf16(load_frag(a1p + kk), b, c1);
      }
      acc0[j] = c0; acc1[j] = c1;
    }
    __syncthreads();
  }

  // final blend: alpha*shared + (1-alpha)*routed
  for (int j = 0; j < 6; ++j) {
    int nt = wave + j * 8;
#pragma unroll
    for (int r = 0; r < 8; ++r) {
      int m0 = mbase + r;
      int m1 = 16 + mbase + r;
      long o0 = (long)(tile * 32 + m0) * kD + nt * 16 + nlane;
      long o1 = (long)(tile * 32 + m1) * kD + nt * 16 + nlane;
      out[o0] = aS[m0] * acc0[j][r] + (1.f - aS[m0]) * routed[o0];
      out[o1] = aS[m1] * acc1[j][r] + (1.f - aS[m1]) * routed[o1];
    }
  }
}

// ---------------- launch ----------------

extern "C" void kernel_launch(void* const* d_in, const int* in_sizes, int n_in,
                              void* d_out, int out_size, void* d_ws, size_t ws_size,
                              hipStream_t stream) {
  (void)in_sizes; (void)n_in; (void)out_size; (void)ws_size;
  const float* x    = (const float*)d_in[0];
  const float* rw   = (const float*)d_in[1];
  const float* wg   = (const float*)d_in[2];
  const float* wu   = (const float*)d_in[3];
  const float* wd   = (const float*)d_in[4];
  const float* wsg  = (const float*)d_in[5];
  const float* wsu  = (const float*)d_in[6];
  const float* wsd  = (const float*)d_in[7];
  const float* sgw  = (const float*)d_in[8];
  const float* sgb  = (const float*)d_in[9];
  float* out = (float*)d_out;

  char* p = (char*)d_ws;
  auto alloc = [&](size_t bytes) -> void* {
    void* r = (void*)p;
    p += (bytes + 255) & ~(size_t)255;
    return r;
  };
  bf16_t* xbf   = (bf16_t*)alloc((size_t)kN * kD * 2);
  bf16_t* wgT   = (bf16_t*)alloc((size_t)kE * kD * kH * 2);
  bf16_t* wuT   = (bf16_t*)alloc((size_t)kE * kD * kH * 2);
  bf16_t* wdT   = (bf16_t*)alloc((size_t)kE * kH * kD * 2);
  bf16_t* wsgT  = (bf16_t*)alloc((size_t)kD * kHS * 2);
  bf16_t* wsuT  = (bf16_t*)alloc((size_t)kD * kHS * 2);
  bf16_t* wsdT  = (bf16_t*)alloc((size_t)kHS * kD * 2);
  float*  routed= (float*)alloc((size_t)kN * kD * 4);
  int*    counts= (int*)alloc(kE * 4);
  int*    bases = (int*)alloc(kE * 4);
  int*    idx   = (int*)alloc((size_t)kN * 2 * 4);
  int*    rank  = (int*)alloc((size_t)kN * 2 * 4);
  float*  gates = (float*)alloc((size_t)kN * 2 * 4);
  float*  alphaA= (float*)alloc((size_t)kN * 4);
  int*    perm  = (int*)alloc((size_t)kN * 2 * 4);
  float*  pgate = (float*)alloc((size_t)kN * 2 * 4);

  moe_zero_kernel<<<(kN * kD + 255) / 256, 256, 0, stream>>>(routed, kN * kD, counts);

  moe_convf2bf_kernel<<<2048, 256, 0, stream>>>(x, xbf, (long)kN * kD);
  moe_tconv_kernel<<<2048, 256, 0, stream>>>(wg,  wgT,  kD,  kH,  kE); // [E][D][H]->[E][H][D]
  moe_tconv_kernel<<<2048, 256, 0, stream>>>(wu,  wuT,  kD,  kH,  kE);
  moe_tconv_kernel<<<2048, 256, 0, stream>>>(wd,  wdT,  kH,  kD,  kE); // [E][H][D]->[E][D][H]
  moe_tconv_kernel<<<2048, 256, 0, stream>>>(wsg, wsgT, kD,  kHS, 1);  // [D][HS]->[HS][D]
  moe_tconv_kernel<<<2048, 256, 0, stream>>>(wsu, wsuT, kD,  kHS, 1);
  moe_tconv_kernel<<<2048, 256, 0, stream>>>(wsd, wsdT, kHS, kD,  1);  // [HS][D]->[D][HS]

  moe_router_kernel<<<(kN * 32 + 255) / 256, 256, 0, stream>>>(
      x, rw, sgw, sgb, idx, gates, rank, alphaA, counts);

  moe_scan_kernel<<<1, 32, 0, stream>>>(counts, bases);
  moe_scatter_kernel<<<(kN + 255) / 256, 256, 0, stream>>>(idx, gates, rank, bases,
                                                           perm, pgate);

  moe_expert_kernel<<<dim3(kE, 32), 256, 0, stream>>>(
      xbf, wgT, wuT, wdT, counts, bases, perm, pgate, routed);

  moe_shared_kernel<<<kN / 32, 256, 0, stream>>>(xbf, wsgT, wsuT, wsdT, alphaA,
                                                 routed, out);
}
